// DVE_loss_40527311405629
// MI455X (gfx1250) — matrix operand order repaired and checked by reading, not compiled, not run on hardware
//
#include <hip/hip_runtime.h>
#include <hip/hip_bf16.h>
#include <math.h>

typedef __attribute__((ext_vector_type(16))) _Float16 v16h;
typedef __attribute__((ext_vector_type(8)))  _Float16 v8h;
typedef __attribute__((ext_vector_type(8)))  float    v8f;

// ---------------------------------------------------------------------------
// Normalize: f1 = l2norm(feats[0::2])*20, f2 = l2norm(feats[1::2])*20 -> f16
// grid: B*N blocks, 128 threads (one per feature dim)
// ---------------------------------------------------------------------------
__global__ void __launch_bounds__(128)
normalize_kernel(const float* __restrict__ feats,
                 _Float16* __restrict__ f1h, _Float16* __restrict__ f2h,
                 int B, int N, int D) {
  int idx = blockIdx.x;
  int b = idx / N, n = idx % N;
  int t = threadIdx.x;                       // t in [0,128) == D
  const float* r1 = feats + ((size_t)(2 * b) * N + n) * D;
  const float* r2 = feats + ((size_t)(2 * b + 1) * N + n) * D;
  float v1 = r1[t], v2 = r2[t];
  __shared__ float s1[128], s2[128];
  s1[t] = v1 * v1; s2[t] = v2 * v2;
  __syncthreads();
  for (int o = 64; o > 0; o >>= 1) {
    if (t < o) { s1[t] += s1[t + o]; s2[t] += s2[t + o]; }
    __syncthreads();
  }
  float inv1 = 20.0f / fmaxf(sqrtf(s1[0]), 1e-12f);
  float inv2 = 20.0f / fmaxf(sqrtf(s2[0]), 1e-12f);
  size_t o = ((size_t)b * N + n) * (size_t)D + t;
  f1h[o] = (_Float16)(v1 * inv1);
  f2h[o] = (_Float16)(v2 * inv2);
}

// ---------------------------------------------------------------------------
// NT GEMM via WMMA: C[MxN] f32 = A[MxK] @ B[NxK]^T, A/B f16 row-major, K%32==0
// One wave computes a 16x64 tile (4 accumulators, A fragment reused).
// Fragment layouts per CDNA5 ISA 7.12.2 (16-bit A 16x32; f32 C/D 16x16).
// ---------------------------------------------------------------------------
__global__ void __launch_bounds__(32)
gemm_nt_kernel(float* __restrict__ C, const _Float16* __restrict__ A,
               const _Float16* __restrict__ Bm, int N, int K, float scale) {
  int lane = threadIdx.x;
  int g = lane >> 4, lr = lane & 15;
  int row0 = blockIdx.y * 16;
  int col0 = blockIdx.x * 64;
  v8f acc[4] = {};
  for (int kb = 0; kb < K; kb += 32) {
    // A fragment: lane holds row (row0+lr); K chunks {kb+g*8..+7, kb+16+g*8..+7}
    const _Float16* pa = A + (size_t)(row0 + lr) * K + kb;
    v8h lo = *(const v8h*)(pa + g * 8);
    v8h hi = *(const v8h*)(pa + 16 + g * 8);
    v16h a;
#pragma unroll
    for (int j = 0; j < 8; ++j) { a[j] = lo[j]; a[j + 8] = hi[j]; }
#pragma unroll
    for (int t = 0; t < 4; ++t) {
      // B fragment (B^T): lane holds column (col0+t*16+lr) = row of Bm,
      // K chunks {kb+g*8..+7, kb+16+g*8..+7}  (16B aligned)
      const _Float16* pb = Bm + (size_t)(col0 + t * 16 + lr) * K + kb;
      v8h blo = *(const v8h*)(pb + g * 8);
      v8h bhi = *(const v8h*)(pb + 16 + g * 8);
      v16h bf;
#pragma unroll
      for (int j = 0; j < 8; ++j) { bf[j] = blo[j]; bf[j + 8] = bhi[j]; }
      acc[t] = __builtin_amdgcn_wmma_f32_16x16x32_f16(
          false, a, false, bf, (short)0, acc[t], false, false);
    }
  }
#pragma unroll
  for (int t = 0; t < 4; ++t)
#pragma unroll
    for (int r = 0; r < 8; ++r)
      C[(size_t)(row0 + r + 8 * g) * N + col0 + t * 16 + lr] = acc[t][r] * scale;
}

// ---------------------------------------------------------------------------
// NN GEMM via WMMA: Out[MxNc] f16 = P[MxK] f16 @ V[KxNc] f16 (f32 accumulate)
// One wave computes a 16x16 tile; used for (softmax @ features), Nc = 128.
// ---------------------------------------------------------------------------
__global__ void __launch_bounds__(32)
gemm_nn_kernel(_Float16* __restrict__ Out, const _Float16* __restrict__ P,
               const _Float16* __restrict__ V, int Nc, int K) {
  int lane = threadIdx.x;
  int g = lane >> 4, lr = lane & 15;
  int row0 = blockIdx.y * 16;
  int col0 = blockIdx.x * 16;
  v8f acc = {};
  for (int kb = 0; kb < K; kb += 32) {
    const _Float16* pa = P + (size_t)(row0 + lr) * K + kb;
    v8h lo = *(const v8h*)(pa + g * 8);
    v8h hi = *(const v8h*)(pa + 16 + g * 8);
    v16h a;
#pragma unroll
    for (int j = 0; j < 8; ++j) { a[j] = lo[j]; a[j + 8] = hi[j]; }
    v16h bf;
#pragma unroll
    for (int j = 0; j < 16; ++j)
      bf[j] = V[(size_t)(kb + (j < 8 ? g * 8 + j : 16 + g * 8 + (j - 8))) * Nc + col0 + lr];
    acc = __builtin_amdgcn_wmma_f32_16x16x32_f16(
        false, a, false, bf, (short)0, acc, false, false);
  }
#pragma unroll
  for (int r = 0; r < 8; ++r)
    Out[(size_t)(row0 + r + 8 * g) * Nc + col0 + lr] = (_Float16)acc[r];
}

// ---------------------------------------------------------------------------
// Row softmax of f32 logits -> f16 probabilities. One block (256) per row.
// ---------------------------------------------------------------------------
__global__ void __launch_bounds__(256)
softmax_row_f16(_Float16* __restrict__ P, const float* __restrict__ C, int N) {
  int row = blockIdx.x, t = threadIdx.x;
  const float* r = C + (size_t)row * N;
  __shared__ float red[256];
  float m = -3.4e38f;
  for (int j = t; j < N; j += 256) m = fmaxf(m, r[j]);
  red[t] = m; __syncthreads();
  for (int o = 128; o > 0; o >>= 1) { if (t < o) red[t] = fmaxf(red[t], red[t + o]); __syncthreads(); }
  m = red[0]; __syncthreads();
  float s = 0.f;
  for (int j = t; j < N; j += 256) s += __expf(r[j] - m);
  red[t] = s; __syncthreads();
  for (int o = 128; o > 0; o >>= 1) { if (t < o) red[t] += red[t + o]; __syncthreads(); }
  float inv = 1.0f / red[0];
  _Float16* p = P + (size_t)row * N;
  for (int j = t; j < N; j += 256) p[j] = (_Float16)(__expf(r[j] - m) * inv);
}

// S = A * alpha (elementwise)
__global__ void scale_kernel(float* __restrict__ S, const float* __restrict__ A,
                             float alpha, size_t n) {
  size_t i = (size_t)blockIdx.x * 256 + threadIdx.x;
  if (i < n) S[i] = A[i] * alpha;
}

// Sinkhorn row pass: S[row,:] -= logsumexp(S[row,:])
__global__ void __launch_bounds__(256)
sink_row_kernel(float* __restrict__ S, int N) {
  int row = blockIdx.x, t = threadIdx.x;
  float* r = S + (size_t)row * N;
  __shared__ float red[256];
  float m = -3.4e38f;
  for (int j = t; j < N; j += 256) m = fmaxf(m, r[j]);
  red[t] = m; __syncthreads();
  for (int o = 128; o > 0; o >>= 1) { if (t < o) red[t] = fmaxf(red[t], red[t + o]); __syncthreads(); }
  m = red[0]; __syncthreads();
  float s = 0.f;
  for (int j = t; j < N; j += 256) s += __expf(r[j] - m);
  red[t] = s; __syncthreads();
  for (int o = 128; o > 0; o >>= 1) { if (t < o) red[t] += red[t + o]; __syncthreads(); }
  float lse = m + __logf(red[0]);
  for (int j = t; j < N; j += 256) r[j] -= lse;
}

// Sinkhorn col pass: one thread owns one column; rows streamed (coalesced
// across adjacent threads), online logsumexp (L2-resident since sequential
// per-sample processing keeps the 16MB matrix hot in the 192MB L2).
__global__ void __launch_bounds__(256)
sink_col_kernel(float* __restrict__ S, int N) {
  int col = blockIdx.x * 256 + threadIdx.x;
  float m = -3.4e38f, s = 0.f;
  for (int r = 0; r < N; ++r) {
    float x = S[(size_t)r * N + col];
    if (x > m) { s = s * __expf(m - x) + 1.0f; m = x; }
    else       { s += __expf(x - m); }
  }
  float lse = m + __logf(s);
  for (int r = 0; r < N; ++r) S[(size_t)r * N + col] -= lse;
}

// ---------------------------------------------------------------------------
// Fused per-row reduction for one sample:
//  loss  += sum_j dist_ij * (softmax(A)_ij + softmax(C12)_ij) / N
//  Lc    += 3*sum_j |exp(S)_ij - softmax(A)_ij| / N
//  perm  += 3*sum_j |[i==j] - exp(S)_ij| / N
//  d     += -diag(softmax(C2))_i / N
//  match += [argmax_j A_ij == i]
// ---------------------------------------------------------------------------
__global__ void __launch_bounds__(256)
reduce_kernel(const float* __restrict__ A, const float* __restrict__ S,
              const float* __restrict__ C12, const float* __restrict__ C2,
              const float* __restrict__ x1, float* __restrict__ acc, int N) {
  int i = blockIdx.x, t = threadIdx.x;
  __shared__ float sx[2048 * 3];
  __shared__ float red[256], red2[256], red3[256];
  __shared__ int   redi[256];
  for (int j = t; j < N * 3; j += 256) sx[j] = x1[j];
  __syncthreads();
  float xi0 = sx[i * 3], xi1 = sx[i * 3 + 1], xi2 = sx[i * 3 + 2];
  float sqi = xi0 * xi0 + xi1 * xi1 + xi2 * xi2;
  const float* rA  = A   + (size_t)i * N;
  const float* rS  = S   + (size_t)i * N;
  const float* r12 = C12 + (size_t)i * N;
  const float* r2  = C2  + (size_t)i * N;
  // phase 1: row maxes + first-argmax of A
  float m1 = -3.4e38f, m2 = -3.4e38f, m3 = -3.4e38f, bv = -3.4e38f;
  int bi = 0;
  for (int j = t; j < N; j += 256) {
    float a = rA[j];
    if (a > bv) { bv = a; bi = j; }
    m1 = fmaxf(m1, a);
    m2 = fmaxf(m2, r12[j]);
    m3 = fmaxf(m3, r2[j]);
  }
  red[t] = m1; red2[t] = m2; red3[t] = m3; __syncthreads();
  for (int o = 128; o > 0; o >>= 1) {
    if (t < o) {
      red[t]  = fmaxf(red[t],  red[t + o]);
      red2[t] = fmaxf(red2[t], red2[t + o]);
      red3[t] = fmaxf(red3[t], red3[t + o]);
    }
    __syncthreads();
  }
  m1 = red[0]; m2 = red2[0]; m3 = red3[0]; __syncthreads();
  red[t] = bv; redi[t] = bi; __syncthreads();
  for (int o = 128; o > 0; o >>= 1) {
    if (t < o) {
      float ov = red[t + o]; int oi = redi[t + o];
      if (ov > red[t] || (ov == red[t] && oi < redi[t])) { red[t] = ov; redi[t] = oi; }
    }
    __syncthreads();
  }
  int amax = redi[0]; __syncthreads();
  // phase 2: sum of exponentials
  float s1 = 0.f, s2 = 0.f, s3 = 0.f;
  for (int j = t; j < N; j += 256) {
    s1 += __expf(rA[j] - m1);
    s2 += __expf(r12[j] - m2);
    s3 += __expf(r2[j] - m3);
  }
  red[t] = s1; red2[t] = s2; red3[t] = s3; __syncthreads();
  for (int o = 128; o > 0; o >>= 1) {
    if (t < o) { red[t] += red[t + o]; red2[t] += red2[t + o]; red3[t] += red3[t + o]; }
    __syncthreads();
  }
  float inv1 = 1.f / red[0], inv2 = 1.f / red2[0], inv3 = 1.f / red3[0];
  __syncthreads();
  // phase 3: weighted sums
  float lsum = 0.f, lc = 0.f, perm = 0.f;
  for (int j = t; j < N; j += 256) {
    float xj0 = sx[j * 3], xj1 = sx[j * 3 + 1], xj2 = sx[j * 3 + 2];
    float dot = xi0 * xj0 + xi1 * xj1 + xi2 * xj2;
    float sqj = xj0 * xj0 + xj1 * xj1 + xj2 * xj2;
    float d2 = fmaxf(sqi + sqj - 2.f * dot, 0.f);
    float dist = sqrtf(sqrtf(d2));            // d2^(POW/2) with POW=0.5
    float p1 = __expf(rA[j] - m1) * inv1;
    float p2 = __expf(r12[j] - m2) * inv2;
    float sk = __expf(rS[j]);
    lsum += dist * (p1 + p2);
    lc   += fabsf(sk - p1);
    perm += fabsf(((j == i) ? 1.f : 0.f) - sk);
  }
  red[t] = lsum; red2[t] = lc; red3[t] = perm; __syncthreads();
  for (int o = 128; o > 0; o >>= 1) {
    if (t < o) { red[t] += red[t + o]; red2[t] += red2[t + o]; red3[t] += red3[t + o]; }
    __syncthreads();
  }
  if (t == 0) {
    float invN = 1.0f / (float)N;
    float pii = __expf(r2[i] - m3) * inv3;    // diag of softmax(corr2) row i
    atomicAdd(&acc[0], red[0] * invN);        // loss_b contribution
    atomicAdd(&acc[1], 3.f * red2[0] * invN); // Lc_b
    atomicAdd(&acc[2], 3.f * red3[0] * invN); // perm_to_I_b
    atomicAdd(&acc[3], -pii * invN);          // d_b
    atomicAdd(&acc[4], (amax == i) ? 1.f : 0.f); // match_b
  }
}

__global__ void zero_kernel(float* acc) { if (threadIdx.x < 8) acc[threadIdx.x] = 0.f; }

__global__ void finalize_kernel(const float* __restrict__ acc, float* __restrict__ out,
                                float invB) {
  if (threadIdx.x == 0) {
    float loss = acc[0], Lc = acc[1], perm = acc[2], d = acc[3], match = acc[4];
    float total = loss + 1.0f * Lc;           // LAMBDA_LC = 1
    out[0] = total * invB;
    out[1] = loss * invB;
    out[2] = Lc * invB;
    out[3] = perm * invB;
    out[4] = match * invB;
    out[5] = d * invB;
  }
}

// ---------------------------------------------------------------------------
extern "C" void kernel_launch(void* const* d_in, const int* in_sizes, int n_in,
                              void* d_out, int out_size, void* d_ws, size_t ws_size,
                              hipStream_t stream) {
  (void)in_sizes; (void)n_in; (void)out_size; (void)ws_size;
  const float* feats = (const float*)d_in[0];   // (16, 2048, 128) f32
  const float* pc0   = (const float*)d_in[1];   // (8, 2048, 3) f32
  // d_in[2] = epoch (unused)
  const int B = 8, N = 2048, D = 128;

  char* ws = (char*)d_ws;
  size_t off = 0;
  auto alloc = [&](size_t bytes) -> void* {
    void* p = ws + off;
    off += (bytes + 255) & ~(size_t)255;
    return p;
  };
  float*     acc   = (float*)alloc(8 * sizeof(float));
  _Float16*  f1h   = (_Float16*)alloc((size_t)B * N * D * 2);
  _Float16*  f2h   = (_Float16*)alloc((size_t)B * N * D * 2);
  _Float16*  f1via = (_Float16*)alloc((size_t)N * D * 2);
  _Float16*  fv16  = (_Float16*)alloc((size_t)N * D * 2);
  float*     T1    = (float*)alloc((size_t)N * N * 4);   // scratch logits (reused as corr2)
  _Float16*  P     = (_Float16*)alloc((size_t)N * N * 2);
  float*     A     = (float*)alloc((size_t)N * N * 4);   // corr_1a2 logits
  float*     S     = (float*)alloc((size_t)N * N * 4);   // sinkhorn log matrix
  float*     C12   = (float*)alloc((size_t)N * N * 4);   // f1@f2.T logits

  zero_kernel<<<1, 32, 0, stream>>>(acc);
  normalize_kernel<<<B * N, 128, 0, stream>>>(feats, f1h, f2h, B, N, D);

  dim3 gNT(N / 64, N / 16);   // NT GEMM: 16x64 tile per wave
  dim3 gNN(D / 16, N / 16);   // NN GEMM: 16x16 tile per wave
  size_t nn = (size_t)N * N;

  for (int b = 0; b < B; ++b) {
    const _Float16* f1b = f1h + (size_t)b * N * D;
    const _Float16* f2b = f2h + (size_t)b * N * D;
    const _Float16* fab = f1h + (size_t)((b + 1) % B) * N * D;   // roll(f1,-1)
    const float*    x1b = pc0 + (size_t)b * N * 3;

    // corr_1a = f1 @ fa.T ; smcorr_1a = softmax ; f1via = smcorr_1a @ fa
    gemm_nt_kernel<<<gNT, 32, 0, stream>>>(T1, f1b, fab, N, D, 1.0f);
    softmax_row_f16<<<N, 256, 0, stream>>>(P, T1, N);
    gemm_nn_kernel<<<gNN, 32, 0, stream>>>(f1via, P, fab, D, N);

    // corr_1a2 and f1@f2.T logits
    gemm_nt_kernel<<<gNT, 32, 0, stream>>>(A, f1via, f2b, N, D, 1.0f);
    gemm_nt_kernel<<<gNT, 32, 0, stream>>>(C12, f1b, f2b, N, D, 1.0f);

    // Sinkhorn: S = A / tau; 30x (row lse-norm, col lse-norm)
    scale_kernel<<<(unsigned)((nn + 255) / 256), 256, 0, stream>>>(S, A, 1.0f / 0.3f, nn);
    for (int it = 0; it < 30; ++it) {
      sink_row_kernel<<<N, 256, 0, stream>>>(S, N);
      sink_col_kernel<<<N / 256, 256, 0, stream>>>(S, N);
    }

    // g = f1@f1.T ; smg ; fv = smg@f1 ; corr2 = fv@f1.T (reuse T1)
    gemm_nt_kernel<<<gNT, 32, 0, stream>>>(T1, f1b, f1b, N, D, 1.0f);
    softmax_row_f16<<<N, 256, 0, stream>>>(P, T1, N);
    gemm_nn_kernel<<<gNN, 32, 0, stream>>>(fv16, P, f1b, D, N);
    gemm_nt_kernel<<<gNT, 32, 0, stream>>>(T1, fv16, f1b, N, D, 1.0f);

    // Fused per-row reduction into accumulators
    reduce_kernel<<<N, 256, 0, stream>>>(A, S, C12, T1, x1b, acc, N);
  }

  finalize_kernel<<<1, 32, 0, stream>>>(acc, (float*)d_out, 1.0f / (float)B);
}
